// SS2Dv2_33921651703902
// MI455X (gfx1250) — compile-verified, matrix operand order checked by reading, not run on hardware
//
#include <hip/hip_runtime.h>
#include <hip/hip_bf16.h>
#include <math.h>

// ---------------- dims (fixed by reference setup) ----------------
#define Bn     4
#define Hn     64
#define Wn     64
#define Ln     4096            // H*W
#define BLn    16384           // B*L
#define Cm     96              // d_model
#define Din    192             // d_inner
#define Nst    16              // d_state
#define Rdt    6               // dt_rank
#define Kdir   4
#define Crows  38              // dt_rank + 2*d_state
#define CrowsP 48              // padded to 3 WMMA tiles
#define CatC   576             // 3*Din
#define Tc     64              // scan timesteps staged per LDS round

typedef __attribute__((ext_vector_type(16))) __bf16 v16bf;
typedef __attribute__((ext_vector_type(8)))  float  v8f;

// packed-pair K base for 16-bit WMMA A/B lane layout (ISA 7.12.2)
__device__ __forceinline__ int wmma_kbase(int j, int half) {
  return (j < 4) ? (half * 8 + 2 * j) : (16 + half * 8 + 2 * (j - 4));
}

__device__ __forceinline__ v8f wmma_bf16(v16bf a, v16bf b, v8f c) {
  return __builtin_amdgcn_wmma_f32_16x16x32_bf16(false, a, false, b, (short)0, c,
                                                 false, false);
}

// direction index map (all four are involutions: gather(u) and scatter(y) use it)
__device__ __forceinline__ int dirmap(int k, int l) {
  switch (k & 3) {
    case 0:  return l;
    case 1:  return ((l & 63) << 6) | (l >> 6);
    case 2:  return (Ln - 1) - l;
    default: { int l2 = (Ln - 1) - l; return ((l2 & 63) << 6) | (l2 >> 6); }
  }
}

__device__ __forceinline__ float silu_f(float v) {
  return v / (1.f + __expf(-v));
}

// ---------------- Kernel 1: in_proj GEMM  xz(16384x384) = x(16384x96) * W^T ----------------
__global__ void k_gemm_inproj(const float* __restrict__ x,
                              const float* __restrict__ w,   // (384,96)
                              float* __restrict__ xz) {
  const int Ndim = 2 * Din, KD = Cm, TN = Ndim / 16;        // 384, 96, 24
  int wave = blockIdx.x * (blockDim.x >> 5) + (threadIdx.x >> 5);
  int lane = threadIdx.x & 31, row = lane & 15, half = lane >> 4;
  int tm = wave / TN, tn = wave % TN;
  if (tm >= BLn / 16) return;
  int m0 = tm * 16, n0 = tn * 16;
  v8f acc = {};
  for (int k0 = 0; k0 < KD; k0 += 32) {
    v16bf a, b;
#pragma unroll
    for (int j = 0; j < 8; ++j) {
      int kb = wmma_kbase(j, half);
      a[2 * j]     = (__bf16)x[(size_t)(m0 + row) * KD + k0 + kb];
      a[2 * j + 1] = (__bf16)x[(size_t)(m0 + row) * KD + k0 + kb + 1];
      b[2 * j]     = (__bf16)w[(size_t)(n0 + row) * KD + k0 + kb];
      b[2 * j + 1] = (__bf16)w[(size_t)(n0 + row) * KD + k0 + kb + 1];
    }
    acc = wmma_bf16(a, b, acc);
  }
#pragma unroll
  for (int r = 0; r < 8; ++r) {
    int mrow = r + half * 8;
    xz[(size_t)(m0 + mrow) * Ndim + n0 + row] = acc[r];
  }
}

// ---------------- Kernel 2: 3x3 depthwise conv (dil=1) + bias + SiLU -> xicT (B,L,Din) ----------------
// block = 192 threads (one per channel d), grid = B*L blocks; all accesses lane-contiguous in d.
__global__ void k_dwconv1(const float* __restrict__ xz,
                          const float* __restrict__ cw,   // (192,1,3,3)
                          const float* __restrict__ cb,
                          float* __restrict__ xicT) {
  int d = threadIdx.x;
  int l = blockIdx.x & (Ln - 1);
  int b = blockIdx.x >> 12;
  int h = l >> 6, w0 = l & 63;
  float s = cb[d];
#pragma unroll
  for (int dy = -1; dy <= 1; ++dy)
#pragma unroll
    for (int dx = -1; dx <= 1; ++dx) {
      int hh = h + dy, ww = w0 + dx;
      if (hh >= 0 && hh < Hn && ww >= 0 && ww < Wn)
        s += xz[((size_t)b * Ln + hh * Wn + ww) * (2 * Din) + d] *
             cw[d * 9 + (dy + 1) * 3 + (dx + 1)];
    }
  xicT[((size_t)b * Ln + l) * Din + d] = silu_f(s);
}

// ---------------- Kernel 3: x_dbl GEMM per (b,k): (48x4096) = Apad(48x192)*Bdir(192x4096) ----------------
// B-operand K-dim (d) is now contiguous per lane (xicT position-major) -> vectorizable loads.
__global__ void k_gemm_xdbl(const float* __restrict__ xicT,
                            const float* __restrict__ xpw,  // (4,38,192)
                            float* __restrict__ xdbl) {     // (B,K,48,L)
  int bk = blockIdx.y, b = bk >> 2, k = bk & 3;
  const int TN = Ln / 16;                                   // 256
  int wave = blockIdx.x * (blockDim.x >> 5) + (threadIdx.x >> 5);
  int lane = threadIdx.x & 31, row = lane & 15, half = lane >> 4;
  int tm = wave / TN, tn = wave % TN;
  if (tm >= CrowsP / 16) return;
  int c0 = tm * 16, l0 = tn * 16;
  int lm = dirmap(k, l0 + row);                             // B-col spatial index
  const float* xtb = xicT + ((size_t)b * Ln + lm) * Din;    // contiguous over d
  const float* wA  = xpw + (size_t)k * Crows * Din;
  int cA = c0 + row;
  bool av = (cA < Crows);
  v8f acc = {};
  for (int k0 = 0; k0 < Din; k0 += 32) {
    v16bf a, bv;
#pragma unroll
    for (int j = 0; j < 8; ++j) {
      int kb = wmma_kbase(j, half);
      a[2 * j]     = (__bf16)(av ? wA[(size_t)cA * Din + k0 + kb]     : 0.f);
      a[2 * j + 1] = (__bf16)(av ? wA[(size_t)cA * Din + k0 + kb + 1] : 0.f);
      bv[2 * j]     = (__bf16)xtb[k0 + kb];
      bv[2 * j + 1] = (__bf16)xtb[k0 + kb + 1];
    }
    acc = wmma_bf16(a, bv, acc);
  }
#pragma unroll
  for (int r = 0; r < 8; ++r) {
    int c = c0 + r + half * 8;
    xdbl[((size_t)bk * CrowsP + c) * Ln + l0 + row] = acc[r];
  }
}

// ---------------- Kernel 4: zero fill ----------------
__global__ void k_zero(float* __restrict__ p, int n) {
  int i = blockIdx.x * blockDim.x + threadIdx.x;
  if (i < n) p[i] = 0.f;
}

// ---------------- Kernel 5: selective scan, 64-step LDS staging, fused dt-proj+softplus ----------------
__global__ void k_scan(const float* __restrict__ xdbl,
                       const float* __restrict__ xicT,  // (B,L,Din)
                       const float* __restrict__ dtw,   // (4,192,6)
                       const float* __restrict__ dtb,   // (4,192)
                       const float* __restrict__ alogs, // (768,16)
                       const float* __restrict__ Dsp,   // (768)
                       float* __restrict__ ysum) {      // (B,L,Din)
  int bk = blockIdx.x, b = bk >> 2, k = bk & 3;
  int d = threadIdx.x;                                  // 0..191
  __shared__ float sm[Crows * Tc];                      // 38 rows x 64 timesteps
  float A[Nst], h[Nst];
#pragma unroll
  for (int n = 0; n < Nst; ++n) {
    A[n] = -__expf(alogs[(size_t)(k * Din + d) * Nst + n]);
    h[n] = 0.f;
  }
  float wdt[Rdt];
#pragma unroll
  for (int r = 0; r < Rdt; ++r) wdt[r] = dtw[(k * Din + d) * Rdt + r];
  float bias = dtb[k * Din + d];
  float Dd   = Dsp[k * Din + d];
  const float* xd = xdbl + (size_t)bk * CrowsP * Ln;
  const float* xt = xicT + (size_t)b * Ln * Din;
  float* yb = ysum + (size_t)b * Ln * Din;
  for (int t0 = 0; t0 < Ln; t0 += Tc) {
    // stage 38x64 block of (dt-rank | B | C) rows; coalesced 64-wide segments
    for (int i = d; i < Crows * Tc; i += Din) {
      int c = i >> 6, tt = i & (Tc - 1);
      sm[i] = xd[(size_t)c * Ln + t0 + tt];
    }
    // prefetch next chunk's row starts (global_prefetch_b8)
    if (d < Crows && t0 + Tc < Ln)
      __builtin_prefetch(&xd[(size_t)d * Ln + t0 + Tc], 0, 1);
    __syncthreads();
#pragma unroll 4
    for (int tt = 0; tt < Tc; ++tt) {
      int lm  = dirmap(k, t0 + tt);
      float u = xt[(size_t)lm * Din + d];               // coalesced over d
      float x = bias;
#pragma unroll
      for (int r = 0; r < Rdt; ++r) x += wdt[r] * sm[r * Tc + tt];
      float dt = (x > 20.f) ? x : log1pf(__expf(x));
      float du = dt * u;
      float y  = 0.f;
#pragma unroll
      for (int n = 0; n < Nst; ++n) {
        h[n] = h[n] * __expf(dt * A[n]) + du * sm[(Rdt + n) * Tc + tt];
        y += h[n] * sm[(Rdt + Nst + n) * Tc + tt];
      }
      y += Dd * u;
      atomicAdd(yb + (size_t)lm * Din + d, y);          // coalesced over d
    }
    __syncthreads();
  }
}

// ---------------- Kernel 6: LayerNorm over channels + SiLU(z) gate -> yact (B,L,Din) ----------------
__global__ void k_ln_gate(const float* __restrict__ ysum,
                          const float* __restrict__ xz,
                          const float* __restrict__ g,
                          const float* __restrict__ bb,
                          float* __restrict__ yact) {
  int bl = blockIdx.x;                   // 0..16383
  int d  = threadIdx.x;                  // 0..191
  __shared__ float red[Din];
  __shared__ float stat[2];
  float v = ysum[(size_t)bl * Din + d];
  red[d] = v;
  __syncthreads();
  if (d < 64) red[d] += red[d + 64] + red[d + 128];
  __syncthreads();
  if (d == 0) { float s = 0.f; for (int i = 0; i < 64; ++i) s += red[i]; stat[0] = s / (float)Din; }
  __syncthreads();
  float mu = stat[0], c = v - mu;
  red[d] = c * c;
  __syncthreads();
  if (d < 64) red[d] += red[d + 64] + red[d + 128];
  __syncthreads();
  if (d == 0) { float s = 0.f; for (int i = 0; i < 64; ++i) s += red[i]; stat[1] = s / (float)Din; }
  __syncthreads();
  float yn = c * rsqrtf(stat[1] + 1e-5f) * g[d] + bb[d];
  float z  = silu_f(xz[(size_t)bl * (2 * Din) + Din + d]);
  yact[(size_t)bl * Din + d] = yn * z;
}

// ---------------- Kernel 7: 3 dilated depthwise convs + BN + ReLU -> cat (B,L,576) ----------------
// block = 192 threads (channel d), grid = 3*B*L; every tap read and the store are d-contiguous.
__global__ void k_branch(const float* __restrict__ yact,  // (B,L,Din)
                         const float* __restrict__ brw,   // (3,192,1,3,3)
                         const float* __restrict__ brg,
                         const float* __restrict__ brb,
                         float* __restrict__ cat) {
  int d = threadIdx.x;
  int l = blockIdx.x & (Ln - 1);
  int rest = blockIdx.x >> 12;
  int b = rest & 3, i = rest >> 2;
  int dil = 1 << i;                        // DIL = (1,2,4)
  int h = l >> 6, w0 = l & 63;
  const float* src = yact + (size_t)b * Ln * Din;
  const float* wgt = brw + (size_t)(i * Din + d) * 9;
  float s = 0.f;
#pragma unroll
  for (int dy = -1; dy <= 1; ++dy)
#pragma unroll
    for (int dx = -1; dx <= 1; ++dx) {
      int hh = h + dy * dil, ww = w0 + dx * dil;
      if (hh >= 0 && hh < Hn && ww >= 0 && ww < Wn)
        s += src[(size_t)(hh * Wn + ww) * Din + d] * wgt[(dy + 1) * 3 + (dx + 1)];
    }
  float sc = brg[i * Din + d] * rsqrtf(1.f + 1e-5f);
  s = fmaxf(s * sc + brb[i * Din + d], 0.f);
  cat[((size_t)b * Ln + l) * CatC + i * Din + d] = s;
}

// ---------------- Kernel 8: fuse GEMM (16384x96x576) + BN + ReLU -> NHWC output ----------------
__global__ void k_gemm_fuse(const float* __restrict__ cat,
                            const float* __restrict__ fw,  // (96,576)
                            const float* __restrict__ fg,
                            const float* __restrict__ fb,
                            float* __restrict__ out) {
  const int Ndim = Cm, KD = CatC, TN = Ndim / 16;          // 96, 576, 6
  int wave = blockIdx.x * (blockDim.x >> 5) + (threadIdx.x >> 5);
  int lane = threadIdx.x & 31, row = lane & 15, half = lane >> 4;
  int tm = wave / TN, tn = wave % TN;
  if (tm >= BLn / 16) return;
  int m0 = tm * 16, n0 = tn * 16;
  v8f acc = {};
  for (int k0 = 0; k0 < KD; k0 += 32) {
    v16bf a, b;
#pragma unroll
    for (int j = 0; j < 8; ++j) {
      int kb = wmma_kbase(j, half);
      a[2 * j]     = (__bf16)cat[(size_t)(m0 + row) * KD + k0 + kb];
      a[2 * j + 1] = (__bf16)cat[(size_t)(m0 + row) * KD + k0 + kb + 1];
      b[2 * j]     = (__bf16)fw[(size_t)(n0 + row) * KD + k0 + kb];
      b[2 * j + 1] = (__bf16)fw[(size_t)(n0 + row) * KD + k0 + kb + 1];
    }
    acc = wmma_bf16(a, b, acc);
  }
  int col = n0 + row;
  float sc = fg[col] * rsqrtf(1.f + 1e-5f);
  float bi = fb[col];
#pragma unroll
  for (int r = 0; r < 8; ++r) {
    int mrow = r + half * 8;
    out[(size_t)(m0 + mrow) * Cm + col] = fmaxf(acc[r] * sc + bi, 0.f);
  }
}

// ---------------- host launcher ----------------
extern "C" void kernel_launch(void* const* d_in, const int* in_sizes, int n_in,
                              void* d_out, int out_size, void* d_ws, size_t ws_size,
                              hipStream_t stream) {
  const float* x    = (const float*)d_in[0];
  const float* ipw  = (const float*)d_in[1];
  const float* cw   = (const float*)d_in[2];
  const float* cb   = (const float*)d_in[3];
  const float* xpw  = (const float*)d_in[4];
  const float* dtw  = (const float*)d_in[5];
  const float* dtb  = (const float*)d_in[6];
  const float* alg  = (const float*)d_in[7];
  const float* Dsp  = (const float*)d_in[8];
  const float* ong  = (const float*)d_in[9];
  const float* onb  = (const float*)d_in[10];
  const float* brw  = (const float*)d_in[11];
  const float* brg  = (const float*)d_in[12];
  const float* brb  = (const float*)d_in[13];
  const float* fw   = (const float*)d_in[14];
  const float* fg   = (const float*)d_in[15];
  const float* fb   = (const float*)d_in[16];
  float* out = (float*)d_out;

  // workspace layout (floats)
  float* ws   = (float*)d_ws;
  float* xz   = ws;                                    // B*L*384  = 6,291,456
  float* xicT = xz   + (size_t)BLn * 2 * Din;          // B*L*Din  = 3,145,728
  float* xdbl = xicT + (size_t)BLn * Din;              // B*K*48*L = 3,145,728
  float* ysum = xdbl + (size_t)Bn * Kdir * CrowsP * Ln;// B*L*Din  = 3,145,728
  float* yact = ysum + (size_t)BLn * Din;              // B*L*Din  = 3,145,728
  float* cat  = yact + (size_t)BLn * Din;              // B*L*576  = 9,437,184

  // 1) in_proj GEMM (WMMA bf16): 24576 tiles, 8 waves/block
  k_gemm_inproj<<<(BLn / 16) * (2 * Din / 16) / 8, 256, 0, stream>>>(x, ipw, xz);
  // 2) depthwise conv + SiLU -> position-major activation
  k_dwconv1<<<BLn, Din, 0, stream>>>(xz, cw, cb, xicT);
  // 3) x_dbl batched GEMM (WMMA bf16): 768 waves per (b,k)
  {
    dim3 g((CrowsP / 16) * (Ln / 16) / 8, Bn * Kdir);
    k_gemm_xdbl<<<g, 256, 0, stream>>>(xicT, xpw, xdbl);
  }
  // 4) zero the accumulation buffer (ws is poisoned by harness)
  k_zero<<<(BLn * Din + 255) / 256, 256, 0, stream>>>(ysum, BLn * Din);
  // 5) selective scan, 4 directions accumulated via coalesced f32 atomics
  k_scan<<<Bn * Kdir, Din, 0, stream>>>(xdbl, xicT, dtw, dtb, alg, Dsp, ysum);
  // 6) LayerNorm + SiLU gate
  k_ln_gate<<<BLn, Din, 0, stream>>>(ysum, xz, ong, onb, yact);
  // 7) dilated depthwise branches + BN + ReLU
  k_branch<<<3 * BLn, Din, 0, stream>>>(yact, brw, brg, brb, cat);
  // 8) fuse GEMM (WMMA bf16) + BN + ReLU, writes NHWC output directly
  k_gemm_fuse<<<(BLn / 16) * (Cm / 16) / 8, 256, 0, stream>>>(cat, fw, fg, fb, out);
}
